// InductiveBuNNLayer_51049981280278
// MI455X (gfx1250) — matrix operand order repaired and checked by reading, not compiled, not run on hardware
//
#include <hip/hip_runtime.h>
#include <hip/hip_bf16.h>

// ---------------------------------------------------------------------------
// InductiveBuNN layer for MI455X (gfx1250).
// Heavy op: 10-step Taylor heat diffusion done as dense fp16 WMMA GEMMs
// against an L2-resident 4096x4096 fp16 normalized-adjacency matrix.
// This revision: each wave computes 16 rows x 64 cols with 4 accumulators,
// so every A (Ltilde) fragment is loaded once and reused by 4 WMMAs
// (4x reduction of the dominant 32MB/iteration A-matrix stream).
// ---------------------------------------------------------------------------

typedef _Float16 h16;
typedef __attribute__((ext_vector_type(16))) _Float16 v16h;
typedef __attribute__((ext_vector_type(8)))  float    v8f;

#define N_NODES 4096
#define FEAT    64

// ---------------------------------------------------------------------------
// 1) degree / message-count initialization
// ---------------------------------------------------------------------------
__global__ void init_nodes_kernel(float* deg, float* cnt, int n) {
    int i = blockIdx.x * blockDim.x + threadIdx.x;
    if (i < n) { deg[i] = 1.0f; /* self loop */ cnt[i] = 0.0f; }
}

__global__ void edge_counts_kernel(const int* __restrict__ ei, int E,
                                   float* deg, float* cnt) {
    int e = blockIdx.x * blockDim.x + threadIdx.x;
    if (e < E) {
        int src = ei[e];        // row for heat Laplacian, src for SAGE
        int dst = ei[E + e];    // col for heat Laplacian, dst for SAGE
        atomicAdd(&deg[src], 1.0f);
        atomicAdd(&cnt[dst], 1.0f);
    }
}

__global__ void dis_kernel(const float* __restrict__ deg, float* dis, int n) {
    int i = blockIdx.x * blockDim.x + threadIdx.x;
    if (i < n) {
        float d = deg[i];
        dis[i] = (d > 0.0f) ? rsqrtf(d) : 0.0f;
    }
}

// ---------------------------------------------------------------------------
// 2) SAGEConv: mean aggregation (scatter-add) + linear
// ---------------------------------------------------------------------------
__global__ void sage_aggr_kernel(const float* __restrict__ x,
                                 const int* __restrict__ ei, int E, int F,
                                 float* msum) {
    int t = blockIdx.x * blockDim.x + threadIdx.x;
    if (t < E * F) {
        int e = t / F;
        int f = t - e * F;
        int src = ei[e];
        int dst = ei[E + e];
        atomicAdd(&msum[dst * F + f], x[src * F + f]);
    }
}

__global__ void sage_lin_kernel(const float* __restrict__ msum,
                                const float* __restrict__ cnt,
                                const float* __restrict__ x,
                                const float* __restrict__ Wl,
                                const float* __restrict__ Wr,
                                const float* __restrict__ b,
                                float* __restrict__ h,
                                int n, int Fin, int Fout) {
    int t = blockIdx.x * blockDim.x + threadIdx.x;
    if (t >= n * Fout) return;
    int i = t / Fout;
    int o = t - i * Fout;
    float inv = 1.0f / fmaxf(cnt[i], 1.0f);
    float s = b[o];
    const float* mrow = msum + i * Fin;
    const float* xrow = x + i * Fin;
    const float* wl = Wl + o * Fin;  // Wl is [Fout, Fin] row-major (h @ Wl.T)
    const float* wr = Wr + o * Fin;
    for (int f = 0; f < Fin; ++f)
        s += mrow[f] * inv * wl[f] + xrow[f] * wr[f];
    h[t] = fmaxf(s, 0.0f);
}

// ---------------------------------------------------------------------------
// 3) Cayley map + rotate-to-bundle-frame; initializes term0 and result=term0
// ---------------------------------------------------------------------------
__global__ void cayley_local_kernel(const float* __restrict__ h2,   // [n,16]
                                    const float* __restrict__ Wp,   // [6,16]
                                    const float* __restrict__ bp,   // [6]
                                    const float* __restrict__ x,    // [n,64]
                                    float* __restrict__ Omat,       // [n,16]
                                    float* __restrict__ term,       // [n,64]
                                    float* __restrict__ result,     // [n,64]
                                    int n) {
    int i = blockIdx.x * blockDim.x + threadIdx.x;
    if (i >= n) return;

    // params = h2 @ Wp.T + bp
    float p[6];
    for (int c = 0; c < 6; ++c) {
        float s = bp[c];
        for (int f = 0; f < 16; ++f) s += h2[i * 16 + f] * Wp[c * 16 + f];
        p[c] = s;
    }
    // skew A, upper-triangle row-major: (0,1)(0,2)(0,3)(1,2)(1,3)(2,3)
    float A[4][4] = {{0.f,  p[0],  p[1],  p[2]},
                     {-p[0], 0.f,  p[3],  p[4]},
                     {-p[1], -p[3], 0.f,  p[5]},
                     {-p[2], -p[4], -p[5], 0.f}};
    // Solve (I - A) O = (I + A), Gauss-Jordan with partial pivoting
    float M[4][8];
    for (int r = 0; r < 4; ++r)
        for (int c = 0; c < 4; ++c) {
            float I = (r == c) ? 1.0f : 0.0f;
            M[r][c]     = I - A[r][c];
            M[r][4 + c] = I + A[r][c];
        }
    for (int col = 0; col < 4; ++col) {
        int piv = col;
        float best = fabsf(M[col][col]);
        for (int r = col + 1; r < 4; ++r) {
            float a = fabsf(M[r][col]);
            if (a > best) { best = a; piv = r; }
        }
        if (piv != col)
            for (int c = 0; c < 8; ++c) {
                float tmp = M[col][c]; M[col][c] = M[piv][c]; M[piv][c] = tmp;
            }
        float inv = 1.0f / M[col][col];
        for (int c = 0; c < 8; ++c) M[col][c] *= inv;
        for (int r = 0; r < 4; ++r) {
            if (r == col) continue;
            float f = M[r][col];
            for (int c = 0; c < 8; ++c) M[r][c] -= f * M[col][c];
        }
    }
    float O[4][4];
    for (int r = 0; r < 4; ++r)
        for (int c = 0; c < 4; ++c) {
            O[r][c] = M[r][4 + c];
            Omat[i * 16 + r * 4 + c] = O[r][c];
        }
    // local[b][ii] = sum_j O[j][ii] * xb[b][j]   (einsum nji,nbj->nbi)
    for (int b = 0; b < 16; ++b)
        for (int ii = 0; ii < 4; ++ii) {
            float s = 0.0f;
            for (int j = 0; j < 4; ++j)
                s += O[j][ii] * x[i * 64 + b * 4 + j];
            term[i * 64 + b * 4 + ii]   = s;
            result[i * 64 + b * 4 + ii] = s;
        }
}

// ---------------------------------------------------------------------------
// 4) Dense fp16 normalized adjacency  Ltilde[i][j] = dis_i * adj_ij * dis_j
//    (Ltilde buffer pre-zeroed; duplicate edges write identical values)
// ---------------------------------------------------------------------------
__global__ void build_L_edges_kernel(const int* __restrict__ ei, int E,
                                     const float* __restrict__ dis,
                                     h16* __restrict__ Lt, int n) {
    int e = blockIdx.x * blockDim.x + threadIdx.x;
    if (e < E) {
        int r = ei[e];
        int c = ei[E + e];
        Lt[(size_t)r * n + c] = (h16)(dis[r] * dis[c]);
    }
}

__global__ void build_L_diag_kernel(const float* __restrict__ dis,
                                    h16* __restrict__ Lt, int n) {
    int i = blockIdx.x * blockDim.x + threadIdx.x;
    if (i < n) Lt[(size_t)i * n + i] = (h16)(dis[i] * dis[i]);
}

// ---------------------------------------------------------------------------
// 5) term [n,64] f32 -> termT [64,n] fp16 (K-contiguous B operand for WMMA)
// ---------------------------------------------------------------------------
__global__ void transpose_cast_kernel(const float* __restrict__ term,
                                      h16* __restrict__ termT, int n) {
    int idx = blockIdx.x * blockDim.x + threadIdx.x;
    if (idx < n * FEAT) {
        int node = idx >> 6;
        int f = idx & 63;
        termT[(size_t)f * n + node] = (h16)term[idx];
    }
}

// ---------------------------------------------------------------------------
// 6) One Taylor step, fully fused:
//      S = Ltilde @ term        (v_wmma_f32_16x16x32_f16, K=4096)
//      new_term = coef * (term - S)       (coef = -t/k)
//      result  += new_term
//    Wave layout: 8 waves/block; each wave owns a 16-row x 64-col strip with
//    4 f32 accumulators, so the A fragment is loaded once per K-chunk and
//    reused by 4 WMMAs. Block covers 128 rows; grid = n/128 blocks.
// ---------------------------------------------------------------------------
union FragH { v16h v; uint4 q[2]; };

__global__ __launch_bounds__(256)
void diffuse_step_kernel(const h16* __restrict__ Lt,      // [n,n] fp16
                         const h16* __restrict__ termT,   // [64,n] fp16
                         const float* __restrict__ term,  // [n,64] f32
                         float* __restrict__ newterm,     // [n,64] f32
                         float* __restrict__ result,      // [n,64] f32
                         float coef, int n) {
    const int lane = threadIdx.x & 31;
    const int wave = threadIdx.x >> 5;         // 0..7
    const int row0 = blockIdx.x * 128 + wave * 16;
    const int l16  = lane & 15;
    const int half = lane >> 4;                // 0: lanes 0-15, 1: lanes 16-31

    const int mrow = row0 + l16;               // A-matrix row for this lane

    // A 16x32 and B 32x16 both use contiguous K: lanes 0-15 K 0..15,
    // lanes 16-31 K 16..31, row/column = lane%16, K contiguous per lane.
    const h16* arow = Lt + (size_t)mrow * n + half * 16;
    const h16* b0 = termT + (size_t)(0 * 16 + l16) * n + half * 16;
    const h16* b1 = termT + (size_t)(1 * 16 + l16) * n + half * 16;
    const h16* b2 = termT + (size_t)(2 * 16 + l16) * n + half * 16;
    const h16* b3 = termT + (size_t)(3 * 16 + l16) * n + half * 16;

    v8f acc0 = {}, acc1 = {}, acc2 = {}, acc3 = {};
    for (int k = 0; k < n; k += 32) {
        FragH a, f0, f1, f2, f3;
        a.q[0]  = *(const uint4*)(arow + k);        // K k+half*16   .. +7
        a.q[1]  = *(const uint4*)(arow + k + 8);    // K k+half*16+8 .. +15
        f0.q[0] = *(const uint4*)(b0 + k);
        f0.q[1] = *(const uint4*)(b0 + k + 8);
        f1.q[0] = *(const uint4*)(b1 + k);
        f1.q[1] = *(const uint4*)(b1 + k + 8);
        f2.q[0] = *(const uint4*)(b2 + k);
        f2.q[1] = *(const uint4*)(b2 + k + 8);
        f3.q[0] = *(const uint4*)(b3 + k);
        f3.q[1] = *(const uint4*)(b3 + k + 8);
        acc0 = __builtin_amdgcn_wmma_f32_16x16x32_f16(
            false, a.v, false, f0.v, (short)0, acc0, false, false);
        acc1 = __builtin_amdgcn_wmma_f32_16x16x32_f16(
            false, a.v, false, f1.v, (short)0, acc1, false, false);
        acc2 = __builtin_amdgcn_wmma_f32_16x16x32_f16(
            false, a.v, false, f2.v, (short)0, acc2, false, false);
        acc3 = __builtin_amdgcn_wmma_f32_16x16x32_f16(
            false, a.v, false, f3.v, (short)0, acc3, false, false);
    }

    // C/D layout: VGPR r -> M = r + half*8, N = lane%16 (+16 per tile)
#pragma unroll
    for (int r = 0; r < 8; ++r) {
        int row = row0 + r + half * 8;
        int base = row * FEAT + l16;
        float v0 = acc0[r], v1 = acc1[r], v2 = acc2[r], v3 = acc3[r];
        float nt0 = coef * (term[base +  0] - v0);
        float nt1 = coef * (term[base + 16] - v1);
        float nt2 = coef * (term[base + 32] - v2);
        float nt3 = coef * (term[base + 48] - v3);
        newterm[base +  0] = nt0;  result[base +  0] += nt0;
        newterm[base + 16] = nt1;  result[base + 16] += nt1;
        newterm[base + 32] = nt2;  result[base + 32] += nt2;
        newterm[base + 48] = nt3;  result[base + 48] += nt3;
    }
}

// ---------------------------------------------------------------------------
// 7) Final: mixed = diffused @ W ; back = O @ mixed (per bundle) + bias; ReLU
//    One block per node, 64 threads, staged through LDS.
// ---------------------------------------------------------------------------
__global__ __launch_bounds__(64)
void finalize_kernel(const float* __restrict__ dif,   // [n,64] (result)
                     const float* __restrict__ W,     // [64,64] row-major
                     const float* __restrict__ Omat,  // [n,16]
                     const float* __restrict__ bias,  // [64]
                     float* __restrict__ out, int n) {
    __shared__ float row[FEAT];
    __shared__ float mix[FEAT];
    int i = blockIdx.x;
    int f = threadIdx.x;
    row[f] = dif[i * FEAT + f];
    __syncthreads();
    float s = 0.0f;
    for (int g = 0; g < FEAT; ++g) s += row[g] * W[g * FEAT + f];
    mix[f] = s;
    __syncthreads();
    int b = f >> 2, ii = f & 3;
    float v = bias[f];
    for (int j = 0; j < 4; ++j)
        v += Omat[i * 16 + ii * 4 + j] * mix[b * 4 + j];  // einsum nij,nbj->nbi
    out[i * FEAT + f] = fmaxf(v, 0.0f);
}

// ---------------------------------------------------------------------------
// host-side launcher
// ---------------------------------------------------------------------------
extern "C" void kernel_launch(void* const* d_in, const int* in_sizes, int n_in,
                              void* d_out, int out_size, void* d_ws, size_t ws_size,
                              hipStream_t stream) {
    const float* x    = (const float*)d_in[0];
    const int*   ei   = (const int*)d_in[1];
    const float* Wl1  = (const float*)d_in[2];
    const float* Wr1  = (const float*)d_in[3];
    const float* b1   = (const float*)d_in[4];
    const float* Wl2  = (const float*)d_in[5];
    const float* Wr2  = (const float*)d_in[6];
    const float* b2   = (const float*)d_in[7];
    const float* Wp   = (const float*)d_in[8];
    const float* bp   = (const float*)d_in[9];
    const float* W    = (const float*)d_in[10];
    const float* bias = (const float*)d_in[11];
    float* out = (float*)d_out;

    const int n = in_sizes[0] / FEAT;   // 4096
    const int E = in_sizes[1] / 2;      // 131072

    // ---- workspace carve-up (256B aligned) ----
    size_t off = 0;
    auto carve = [&](size_t bytes) {
        size_t o = off;
        off = (off + bytes + 255) & ~(size_t)255;
        return o;
    };
    char* ws = (char*)d_ws;
    size_t o_Lt    = carve((size_t)n * n * sizeof(h16));     // 32 MB
    size_t o_termA = carve((size_t)n * FEAT * sizeof(float));
    size_t o_termB = carve((size_t)n * FEAT * sizeof(float));
    size_t o_res   = carve((size_t)n * FEAT * sizeof(float));
    size_t o_termT = carve((size_t)FEAT * n * sizeof(h16));
    size_t o_msum  = carve((size_t)n * FEAT * sizeof(float));
    size_t o_cnt   = carve((size_t)n * sizeof(float));
    size_t o_deg   = carve((size_t)n * sizeof(float));
    size_t o_dis   = carve((size_t)n * sizeof(float));
    size_t o_h1    = carve((size_t)n * 32 * sizeof(float));
    size_t o_h2    = carve((size_t)n * 16 * sizeof(float));
    size_t o_O     = carve((size_t)n * 16 * sizeof(float));
    if (off > ws_size) return;  // workspace too small; bail deterministically

    h16*   Lt    = (h16*)(ws + o_Lt);
    float* termA = (float*)(ws + o_termA);
    float* termB = (float*)(ws + o_termB);
    float* resid = (float*)(ws + o_res);
    h16*   termT = (h16*)(ws + o_termT);
    float* msum  = (float*)(ws + o_msum);
    float* cnt   = (float*)(ws + o_cnt);
    float* deg   = (float*)(ws + o_deg);
    float* dis   = (float*)(ws + o_dis);
    float* h1    = (float*)(ws + o_h1);
    float* h2    = (float*)(ws + o_h2);
    float* Omat  = (float*)(ws + o_O);

    const int TB = 256;
    auto blocks = [](long long t, int tb) { return (unsigned)((t + tb - 1) / tb); };

    // degrees & counts
    init_nodes_kernel<<<blocks(n, TB), TB, 0, stream>>>(deg, cnt, n);
    edge_counts_kernel<<<blocks(E, TB), TB, 0, stream>>>(ei, E, deg, cnt);

    // SAGE layer 1: 64 -> 32
    hipMemsetAsync(msum, 0, (size_t)n * 64 * sizeof(float), stream);
    sage_aggr_kernel<<<blocks((long long)E * 64, TB), TB, 0, stream>>>(x, ei, E, 64, msum);
    sage_lin_kernel<<<blocks((long long)n * 32, TB), TB, 0, stream>>>(
        msum, cnt, x, Wl1, Wr1, b1, h1, n, 64, 32);

    // SAGE layer 2: 32 -> 16
    hipMemsetAsync(msum, 0, (size_t)n * 32 * sizeof(float), stream);
    sage_aggr_kernel<<<blocks((long long)E * 32, TB), TB, 0, stream>>>(h1, ei, E, 32, msum);
    sage_lin_kernel<<<blocks((long long)n * 16, TB), TB, 0, stream>>>(
        msum, cnt, h1, Wl2, Wr2, b2, h2, n, 32, 16);

    // Cayley + rotate-to-frame; initializes term0 and result
    cayley_local_kernel<<<blocks(n, TB), TB, 0, stream>>>(
        h2, Wp, bp, x, Omat, termA, resid, n);

    // dense fp16 normalized adjacency (L2-resident)
    hipMemsetAsync(Lt, 0, (size_t)n * n * sizeof(h16), stream);
    dis_kernel<<<blocks(n, TB), TB, 0, stream>>>(deg, dis, n);
    build_L_edges_kernel<<<blocks(E, TB), TB, 0, stream>>>(ei, E, dis, Lt, n);
    build_L_diag_kernel<<<blocks(n, TB), TB, 0, stream>>>(dis, Lt, n);

    // 10 Taylor steps of exp(-L): term <- (-1/k)(term - Lt@term); result += term
    float* ta = termA;
    float* tb = termB;
    for (int k = 1; k <= 10; ++k) {
        transpose_cast_kernel<<<blocks((long long)n * FEAT, TB), TB, 0, stream>>>(ta, termT, n);
        diffuse_step_kernel<<<n / 128, 256, 0, stream>>>(
            Lt, termT, ta, tb, resid, -1.0f / (float)k, n);
        float* tmp = ta; ta = tb; tb = tmp;
    }

    // channel mix + rotate back + bias + relu
    finalize_kernel<<<n, 64, 0, stream>>>(resid, W, Omat, bias, out, n);
}
